// SparseGraphEncoderLayer_31791347925245
// MI455X (gfx1250) — matrix operand order repaired and checked by reading, not compiled, not run on hardware
//
#include <hip/hip_runtime.h>
#include <hip/hip_bf16.h>

// ---------------------------------------------------------------------------
// SparseGraphEncoderLayer for MI455X (gfx1250), bf16 WMMA (f32 accumulate).
// B=8, N=96, D=512, E=4. All GEMMs run on v_wmma_f32_16x16x32_bf16.
// B operands are pre-transposed so both WMMA operands load as contiguous
// 32-byte chunks (2x b128 per lane), no scalar gather/packing.
// ---------------------------------------------------------------------------

typedef __attribute__((ext_vector_type(16))) __bf16 v16bf;
typedef __attribute__((ext_vector_type(8)))  float  v8f;

union BfVec {
    v16bf v;
    unsigned short u[16];
    unsigned long long d[4];
};

__device__ __forceinline__ unsigned short f2bf(float f) {
    unsigned int u = __float_as_uint(f);
    u += 0x7FFFu + ((u >> 16) & 1u);   // round-to-nearest-even
    return (unsigned short)(u >> 16);
}
__device__ __forceinline__ float bf2f(unsigned short h) {
    return __uint_as_float(((unsigned int)h) << 16);
}

__device__ __forceinline__ void load_bfvec(BfVec& x, const unsigned short* p) {
    const unsigned long long* q = (const unsigned long long*)p;
    x.d[0] = q[0]; x.d[1] = q[1]; x.d[2] = q[2]; x.d[3] = q[3];
}

// One wave computes a 16x64 strip (4 x 16x16 tiles) of C = A * B.
// A: bf16 MxK row-major (lda). BT: bf16 NxK row-major (ldbt) == B transposed.
// K % 32 == 0.
// A-operand k order this code uses (B matches it): lanes 0-15 rows M=0..15 hold
// K = {kk..kk+15}; lanes 16-31 hold K = {kk+16..kk+31}.
// B-operand: lane col N=lane%16 (+16 cols handled by halves), 16 consecutive K.
// OUTMODE: 0 = f32 row-major, 1 = bf16 row-major, 2 = bf16 transposed (C[col*ldc+row])
template <int OUTMODE>
__device__ __forceinline__ void wmma_strip(const unsigned short* __restrict__ A, int lda,
                                           const unsigned short* __restrict__ BT, int ldbt,
                                           void* __restrict__ C, int ldc,
                                           int K, int tm, int tn4, int lane)
{
    const int mrow   = lane & 15;
    const int half   = lane >> 4;          // 0 or 1
    const int khalfA = half * 16;
    const int kbaseB = half * 16;

    v8f c[4] = {};
    const unsigned short* arow = A + (long)(tm * 16 + mrow) * lda;
    const unsigned short* brow[4];
#pragma unroll
    for (int q = 0; q < 4; ++q)
        brow[q] = BT + (long)(tn4 * 64 + q * 16 + mrow) * ldbt;

    for (int kk = 0; kk < K; kk += 32) {
        BfVec a;
        // two contiguous 16B chunks: K = {kk+khalfA .. +7} and {kk+8+khalfA .. +7}
        const unsigned long long* p0 = (const unsigned long long*)(arow + kk + khalfA);
        a.d[0] = p0[0]; a.d[1] = p0[1];
        const unsigned long long* p1 = (const unsigned long long*)(arow + kk + khalfA + 8);
        a.d[2] = p1[0]; a.d[3] = p1[1];

        if (kk + 32 < K) {
            __builtin_prefetch(arow + kk + 32, 0, 1);
            __builtin_prefetch(brow[0] + kk + 32, 0, 1);
        }
#pragma unroll
        for (int q = 0; q < 4; ++q) {
            BfVec b;
            load_bfvec(b, brow[q] + kk + kbaseB);   // 16 consecutive K values
            c[q] = __builtin_amdgcn_wmma_f32_16x16x32_bf16(
                       false, a.v, false, b.v, (short)0, c[q], false, false);
        }
    }

    // C/D layout: VGPR r holds M = r (lanes 0-15) / 8+r (lanes 16-31), N = lane%16.
#pragma unroll
    for (int q = 0; q < 4; ++q) {
        int col = tn4 * 64 + q * 16 + mrow;
#pragma unroll
        for (int r2 = 0; r2 < 8; ++r2) {
            int row = tm * 16 + half * 8 + r2;
            if (OUTMODE == 0)
                ((float*)C)[(long)row * ldc + col] = c[q][r2];
            else if (OUTMODE == 1)
                ((unsigned short*)C)[(long)row * ldc + col] = f2bf(c[q][r2]);
            else
                ((unsigned short*)C)[(long)col * ldc + row] = f2bf(c[q][r2]);
        }
    }
}

// ---------------------------------------------------------------------------
// Conversion kernels
// ---------------------------------------------------------------------------
__global__ void f2bf_kernel(const float* __restrict__ src,
                            unsigned short* __restrict__ dst, int n) {
    int i = blockIdx.x * blockDim.x + threadIdx.x;
    if (i < n) dst[i] = f2bf(src[i]);
}

// batched transpose+convert: src[batch][rows][cols] f32 -> dst[batch][cols][rows] bf16
__global__ void f2bf_t_kernel(const float* __restrict__ src,
                              unsigned short* __restrict__ dst,
                              int rows, int cols) {
    int i = blockIdx.x * blockDim.x + threadIdx.x;
    long n = (long)rows * cols;
    long boff = (long)blockIdx.y * n;
    if (i < n) {
        int r = i / cols, cnum = i - r * cols;
        dst[boff + (long)cnum * rows + r] = f2bf(src[boff + i]);
    }
}

// nodes [768,512] f32 -> azb[:, 1024:1536] bf16 (az = [in_h, out_h, nodes])
__global__ void nodes_to_azb_kernel(const float* __restrict__ nodes,
                                    unsigned short* __restrict__ azb) {
    int i = blockIdx.x * blockDim.x + threadIdx.x;
    if (i < 768 * 512) {
        int row = i >> 9, col = i & 511;
        azb[(long)row * 1536 + 1024 + col] = f2bf(nodes[i]);
    }
}

// ---------------------------------------------------------------------------
// K1: M[dir,b,e] (96x512) = nodes[b] (96x512) @ W[dir,e] (512x512).
// Output written TRANSPOSED into MbT[dir,b] (512 x 384): MbT[d][e*96+j].
// grid (12, 64): 48 wave-strips (6 tm x 8 tn4) / 4 waves, batch = dir*32+b*4+e.
// ---------------------------------------------------------------------------
__global__ void __launch_bounds__(128)
msgs_gemm_kernel(const unsigned short* __restrict__ azb,
                 const unsigned short* __restrict__ edgesT,
                 unsigned short* __restrict__ MbT) {
    int lane = threadIdx.x & 31, wave = threadIdx.x >> 5;
    int tile = blockIdx.x * 4 + wave;            // [0,48)
    int tm = tile >> 3, tn4 = tile & 7;          // 6 x 8
    int batch = blockIdx.y;                      // [0,64) = dir*32 + b*4 + e
    int dir = batch >> 5, b = (batch >> 2) & 7, e = batch & 3;
    int db = (dir << 3) | b;
    const unsigned short* A  = azb + (long)(b * 96) * 1536 + 1024;   // nodes slice
    const unsigned short* BT = edgesT + (long)(dir * 4 + e) * 512 * 512;
    unsigned short* C = MbT + (long)db * 512 * 384 + e * 96;
    wmma_strip<2>(A, 1536, BT, 512, C, 384, 512, tm, tn4, lane);
}

// ---------------------------------------------------------------------------
// K2: s1[dir,b,i] = nodes[b,i]·Wa_dir[:512]; s2[db*384 + e*96 + j] =
//     M[dir,b,e,j]·Wa_dir[512:]  (M read from transposed MbT)
// ---------------------------------------------------------------------------
__global__ void __launch_bounds__(128)
dots_kernel(const float* __restrict__ nodes,
            const float* __restrict__ Wa_in, const float* __restrict__ Wa_out,
            const unsigned short* __restrict__ MbT,
            float* __restrict__ s1, float* __restrict__ s2) {
    __shared__ float red[128];
    int id = blockIdx.x;        // [0, 1536 + 6144)
    int t  = threadIdx.x;
    float acc = 0.f;
    if (id < 1536) {
        int dir = id / 768;
        int r   = id - dir * 768;                // b*96 + i
        const float* v = nodes + (long)r * 512;
        const float* w = dir ? Wa_out : Wa_in;
        for (int k = t; k < 512; k += 128) acc += v[k] * w[k];
    } else {
        int id2 = id - 1536;                     // db*384 + e*96 + j
        int db  = id2 / 384;
        int cnum = id2 - db * 384;
        int dir = db >> 3;
        const unsigned short* v = MbT + (long)db * 512 * 384 + cnum;
        const float* w = (dir ? Wa_out : Wa_in) + 512;
        for (int k = t; k < 512; k += 128) acc += bf2f(v[(long)k * 384]) * w[k];
    }
    red[t] = acc;
    __syncthreads();
    for (int s = 64; s > 0; s >>= 1) {
        if (t < s) red[t] += red[t + s];
        __syncthreads();
    }
    if (t == 0) {
        if (id < 1536) s1[id] = red[0];
        else           s2[id - 1536] = red[0];
    }
}

// ---------------------------------------------------------------------------
// K3: per (dir,b,i): score_j = leaky(s1+s2[t,j]+ba) masked, softmax over j,
// scatter into Ahat[dir,b,i, e*96+j] (bf16, zero where type!=e or !exist).
// ---------------------------------------------------------------------------
__global__ void __launch_bounds__(128)
attn_kernel(const int* __restrict__ adj_in, const int* __restrict__ adj_out,
            const unsigned char* __restrict__ ex_in, const unsigned char* __restrict__ ex_out,
            const float* __restrict__ ba_in, const float* __restrict__ ba_out,
            const float* __restrict__ s1, const float* __restrict__ s2,
            unsigned short* __restrict__ Ahat) {
    __shared__ float red[128];
    int id  = blockIdx.x;                  // dir*768 + b*96 + i
    int dir = id / 768;
    int r   = id - dir * 768;
    int b   = r / 96, i = r - b * 96;
    int j   = threadIdx.x;
    bool act = j < 96;

    const int* adj = dir ? adj_out : adj_in;
    const unsigned char* ex = dir ? ex_out : ex_in;
    float bav = dir ? ba_out[0] : ba_in[0];

    int  ty = 0;
    bool exists = false;
    float score = -1e9f;
    if (act) {
        long off = (long)(b * 96 + i) * 96 + j;
        ty = adj[off] & 3;
        exists = ex[off] != 0;
        float x = s1[id] + s2[((dir * 8 + b) * 4 + ty) * 96 + j] + bav;
        x = (x >= 0.f) ? x : 0.2f * x;     // LeakyReLU(0.2)
        score = exists ? x : -1e9f;
    }

    red[threadIdx.x] = act ? score : -1e30f;
    __syncthreads();
    for (int s = 64; s > 0; s >>= 1) {
        if (threadIdx.x < s)
            red[threadIdx.x] = fmaxf(red[threadIdx.x], red[threadIdx.x + s]);
        __syncthreads();
    }
    float mx = red[0];
    __syncthreads();
    float e = act ? __expf(score - mx) : 0.f;
    red[threadIdx.x] = e;
    __syncthreads();
    for (int s = 64; s > 0; s >>= 1) {
        if (threadIdx.x < s) red[threadIdx.x] += red[threadIdx.x + s];
        __syncthreads();
    }
    float attn = e / red[0];

    if (act) {
        long rowbase = (long)id * 384;
#pragma unroll
        for (int e4 = 0; e4 < 4; ++e4) {
            float v = (exists && ty == e4) ? attn : 0.f;
            Ahat[rowbase + e4 * 96 + j] = f2bf(v);
        }
    }
}

// ---------------------------------------------------------------------------
// K4: agg[dir,b] (96x512) = Ahat[dir,b] (96x384) @ M[dir,b] (384x512),
// B operand taken from MbT (512x384 = M^T), written bf16 into azb columns
// [dir*512 : dir*512+512].  grid (12, 16).
// ---------------------------------------------------------------------------
__global__ void __launch_bounds__(128)
agg_gemm_kernel(const unsigned short* __restrict__ Ahat,
                const unsigned short* __restrict__ MbT,
                unsigned short* __restrict__ azb) {
    int lane = threadIdx.x & 31, wave = threadIdx.x >> 5;
    int tile = blockIdx.x * 4 + wave;            // [0,48)
    int tm = tile >> 3, tn4 = tile & 7;
    int db = blockIdx.y;                         // dir*8 + b, [0,16)
    int dir = db >> 3, b = db & 7;
    const unsigned short* A  = Ahat + (long)db * 96 * 384;
    const unsigned short* BT = MbT + (long)db * 512 * 384;   // (D x E*N) = M^T
    unsigned short* C = azb + (long)(b * 96) * 1536 + dir * 512;
    wmma_strip<1>(A, 384, BT, 384, C, 1536, 384, tm, tn4, lane);
}

// ---------------------------------------------------------------------------
// K5: generic [768 x K] @ [K x 512] -> f32 (B given transposed, ldbt elems).
// grid 96, 4 waves -> 48 x 8 wave-strips.
// ---------------------------------------------------------------------------
__global__ void __launch_bounds__(128)
big_gemm_kernel(const unsigned short* __restrict__ A, int lda,
                const unsigned short* __restrict__ BT, int ldbt,
                float* __restrict__ C, int ldc, int K) {
    int lane = threadIdx.x & 31, wave = threadIdx.x >> 5;
    int tile = blockIdx.x * 4 + wave;            // [0,384)
    int tm = tile >> 3, tn4 = tile & 7;
    wmma_strip<0>(A, lda, BT, ldbt, C, ldc, K, tm, tn4, lane);
}

// r = sigmoid(Rpre + br); rnb = bf16(r * nodes)
__global__ void rn_kernel(const float* __restrict__ Rpre, const float* __restrict__ br,
                          const float* __restrict__ nodes, unsigned short* __restrict__ rnb) {
    int i = blockIdx.x * blockDim.x + threadIdx.x;
    if (i < 768 * 512) {
        int d = i & 511;
        float rr = 1.f / (1.f + __expf(-(Rpre[i] + br[d])));
        rnb[i] = f2bf(rr * nodes[i]);
    }
}

// out = (1-z)*nodes + z*tanh(P + Q + bh), z = sigmoid(Zpre + bz)
__global__ void final_kernel(const float* __restrict__ Zpre, const float* __restrict__ bz,
                             const float* __restrict__ P, const float* __restrict__ Q,
                             const float* __restrict__ bh, const float* __restrict__ nodes,
                             float* __restrict__ out) {
    int i = blockIdx.x * blockDim.x + threadIdx.x;
    if (i < 768 * 512) {
        int d = i & 511;
        float z  = 1.f / (1.f + __expf(-(Zpre[i] + bz[d])));
        float hh = tanhf(P[i] + Q[i] + bh[d]);
        out[i] = (1.f - z) * nodes[i] + z * hh;
    }
}

// ---------------------------------------------------------------------------
extern "C" void kernel_launch(void* const* d_in, const int* in_sizes, int n_in,
                              void* d_out, int out_size, void* d_ws, size_t ws_size,
                              hipStream_t stream) {
    (void)in_sizes; (void)n_in; (void)out_size; (void)ws_size;
    const float*         nodes   = (const float*)d_in[0];
    const float*         edges   = (const float*)d_in[1];
    const int*           adj_in  = (const int*)d_in[2];
    const int*           adj_out = (const int*)d_in[3];
    const unsigned char* ex_in   = (const unsigned char*)d_in[4];
    const unsigned char* ex_out  = (const unsigned char*)d_in[5];
    const float*         Wa_in   = (const float*)d_in[6];
    const float*         ba_in   = (const float*)d_in[7];
    const float*         Wa_out  = (const float*)d_in[8];
    const float*         ba_out  = (const float*)d_in[9];
    const float*         Wr      = (const float*)d_in[10];
    const float*         br      = (const float*)d_in[11];
    const float*         Wz      = (const float*)d_in[12];
    const float*         bz      = (const float*)d_in[13];
    const float*         Wh      = (const float*)d_in[14];
    const float*         bh      = (const float*)d_in[15];
    float* out = (float*)d_out;

    char* ws = (char*)d_ws;
    size_t off = 0;
    auto alloc = [&](size_t bytes) -> char* {
        char* p = ws + off;
        off += (bytes + 255) & ~((size_t)255);
        return p;
    };
    unsigned short* azb    = (unsigned short*)alloc(768UL * 1536 * 2);       // [a | nodes] bf16
    unsigned short* edgesT = (unsigned short*)alloc(2UL * 4 * 512 * 512 * 2); // W^T per (dir,e)
    unsigned short* WrT    = (unsigned short*)alloc(512UL * 1536 * 2);
    unsigned short* WzT    = (unsigned short*)alloc(512UL * 1536 * 2);
    unsigned short* WhT    = (unsigned short*)alloc(512UL * 1536 * 2);
    unsigned short* MbT    = (unsigned short*)alloc(16UL * 512 * 384 * 2);   // M^T per (dir,b)
    unsigned short* Ahat   = (unsigned short*)alloc(16UL * 96 * 384 * 2);
    float*          s1     = (float*)alloc(1536UL * 4);
    float*          s2     = (float*)alloc(6144UL * 4);
    float*          Rpre   = (float*)alloc(768UL * 512 * 4);
    float*          Zpre   = (float*)alloc(768UL * 512 * 4);
    float*          Pbuf   = (float*)alloc(768UL * 512 * 4);
    float*          Qbuf   = (float*)alloc(768UL * 512 * 4);
    unsigned short* rnb    = (unsigned short*)alloc(768UL * 512 * 2);

    const int thr = 256;
    // transpose+convert weights to bf16
    f2bf_t_kernel<<<dim3((512 * 512 + thr - 1) / thr, 8), thr, 0, stream>>>(edges, edgesT, 512, 512);
    f2bf_t_kernel<<<dim3((1536 * 512 + thr - 1) / thr, 1), thr, 0, stream>>>(Wr, WrT, 1536, 512);
    f2bf_t_kernel<<<dim3((1536 * 512 + thr - 1) / thr, 1), thr, 0, stream>>>(Wz, WzT, 1536, 512);
    f2bf_t_kernel<<<dim3((1536 * 512 + thr - 1) / thr, 1), thr, 0, stream>>>(Wh, WhT, 1536, 512);
    nodes_to_azb_kernel<<<393216 / thr, thr, 0, stream>>>(nodes, azb);

    msgs_gemm_kernel<<<dim3(12, 64), 128, 0, stream>>>(azb, edgesT, MbT);
    dots_kernel<<<7680, 128, 0, stream>>>(nodes, Wa_in, Wa_out, MbT, s1, s2);
    attn_kernel<<<1536, 128, 0, stream>>>(adj_in, adj_out, ex_in, ex_out,
                                          ba_in, ba_out, s1, s2, Ahat);
    agg_gemm_kernel<<<dim3(12, 16), 128, 0, stream>>>(Ahat, MbT, azb);

    // R = az@Wr, Z = az@Wz, P = a@Wh[:1024]  (WhT row d holds Wh[:,d])
    big_gemm_kernel<<<96, 128, 0, stream>>>(azb, 1536, WrT, 1536, Rpre, 512, 1536);
    big_gemm_kernel<<<96, 128, 0, stream>>>(azb, 1536, WzT, 1536, Zpre, 512, 1536);
    big_gemm_kernel<<<96, 128, 0, stream>>>(azb, 1536, WhT, 1536, Pbuf, 512, 1024);
    rn_kernel<<<393216 / thr, thr, 0, stream>>>(Rpre, br, nodes, rnb);
    // Q = (r*nodes) @ Wh[1024:1536] : B^T = WhT columns 1024.. (row d, offset 1024)
    big_gemm_kernel<<<96, 128, 0, stream>>>(rnb, 512, WhT + 1024, 1536, Qbuf, 512, 512);
    final_kernel<<<393216 / thr, thr, 0, stream>>>(Zpre, bz, Pbuf, Qbuf, bh, nodes, out);
}
